// MambaPFT_7567732375830
// MI455X (gfx1250) — matrix operand order repaired and checked by reading, not compiled, not run on hardware
//
#include <hip/hip_runtime.h>

typedef __attribute__((ext_vector_type(16))) _Float16 v16h;
typedef __attribute__((ext_vector_type(8)))  _Float16 v8h;
typedef __attribute__((ext_vector_type(8)))  float    v8f;
typedef _Float16 h16;

typedef unsigned int u32x4 __attribute__((ext_vector_type(4)));
typedef int          i32x4 __attribute__((ext_vector_type(4)));
typedef int          i32x8 __attribute__((ext_vector_type(8)));

#if defined(__gfx1250__) && __has_builtin(__builtin_amdgcn_tensor_load_to_lds) && \
    __has_builtin(__builtin_amdgcn_s_wait_tensorcnt)
#define USE_TDM 1
#else
#define USE_TDM 0
#endif

#define HW 16384          // 128*128
#define CF 64             // F_FEAT
#define DI 128            // D_INNER
#define DS 16             // D_STATE
#define XZW 416           // 2*DI + 2*DS + DI

// ---------------- device helpers ----------------
__device__ __forceinline__ float sigmoidf_(float x){ return 1.f/(1.f+__expf(-x)); }
__device__ __forceinline__ float siluf_(float x){ return x*sigmoidf_(x); }
__device__ __forceinline__ float softplusf_(float x){ return (x>20.f)?x:log1pf(__expf(x)); }
__device__ __forceinline__ float geluf_(float x){ return 0.5f*x*(1.f+erff(x*0.7071067811865475f)); }
__device__ __forceinline__ float cubicw_(float t){
  const float a=-0.75f; float at=fabsf(t);
  if(at<=1.f) return ((a+2.f)*at-(a+3.f))*at*at+1.f;
  if(at<2.f)  return a*(((at-5.f)*at+8.f)*at-4.f);
  return 0.f;
}
__device__ __forceinline__ v16h cat8_(v8h lo, v8h hi){
  return __builtin_shufflevector(lo, hi, 0,1,2,3,4,5,6,7,8,9,10,11,12,13,14,15);
}

// ---------------- WMMA GEMM ----------------
// C[M][N] f32 = A[M][K]f16(row-major) * Bp (fragment-packed: [(kb*N+n)*32 + kloc])
// grid.x = M/16, grid.y = ceil((N/16)/4), block = 128 (4 waves, one 16x16 tile each)
__global__ void mpft_gemm(const h16* __restrict__ A, const h16* __restrict__ Bp,
                          float* __restrict__ C, int M, int N, int K){
  const int lane = threadIdx.x & 31;
  const int wave = threadIdx.x >> 5;
  const int mBase = blockIdx.x * 16;
  const int nTile = blockIdx.y * 4 + wave;
  if (nTile * 16 >= N) return;           // wave-uniform
  const int nBase = nTile * 16;
  const int row   = mBase + (lane & 15);
  const int kOffA = (lane < 16) ? 0 : 8;   // A: lane<16 -> K{0..7,16..23}
  const int kSelB = (lane < 16) ? 0 : 8;   // B: lane<16 -> K{0..7,16..23}
  const int col   = nBase + (lane & 15);
  v8f acc = {0.f,0.f,0.f,0.f,0.f,0.f,0.f,0.f};
  for (int kb = 0; kb < K; kb += 32) {
    const h16* ap = A + (size_t)row * K + kb + kOffA;
    v16h a = cat8_(*(const v8h*)ap, *(const v8h*)(ap + 16));
    const h16* bp = Bp + ((size_t)(kb >> 5) * N + col) * 32 + kSelB;
    v16h b = cat8_(*(const v8h*)bp, *(const v8h*)(bp + 16));
    if (kb + 32 < K)
      __builtin_prefetch(Bp + ((size_t)((kb >> 5) + 1) * N + col) * 32, 0, 0);
    acc = __builtin_amdgcn_wmma_f32_16x16x32_f16(false, a, false, b,
                                                 (short)0, acc, false, false);
  }
  const int n = nBase + (lane & 15);
  const int mAdd = (lane < 16) ? 0 : 8;
  #pragma unroll
  for (int r = 0; r < 8; ++r)
    C[(size_t)(mBase + r + mAdd) * N + n] = acc[r];
}

// pack f32 [K][N] row-major -> fragment-packed f16 [(kb*N+n)*32 + kloc]
__global__ void mpft_pack_b(const float* __restrict__ B, h16* __restrict__ out, int N, int total){
  int t = blockIdx.x * blockDim.x + threadIdx.x;
  if (t >= total) return;
  int kb = t / (N * 32), rem = t % (N * 32), n = rem / 32, kl = rem & 31;
  out[t] = (h16)B[(size_t)(kb * 32 + kl) * N + n];
}

// ---------------- WMMA 3x3 conv, IC=64 ----------------
// X: f16 HWC [H][W][64]; Wp: f16 packed [((ks*OC)+o)*32 + kloc] (k = tap*64+ic, ks=k/32)
// Y: f32 CHW [OC][H][W]
// grid.x = H*W/16 (16-pixel row strips), grid.y = OC/64, block = 128
__device__ __forceinline__ void conv_stage_manual(const h16* __restrict__ X,
                                                  h16 (*patch)[18][64],
                                                  int y, int x0, int H, int W){
  for (int idx = threadIdx.x; idx < 3*18*64; idx += 128) {
    int dy = idx / (18*64); int r = idx % (18*64);
    int xx = r / 64;        int ic = r % 64;
    int gy = y + dy - 1, gx = x0 + xx - 1;
    h16 v = (h16)0.f;
    if (gy >= 0 && gy < H && gx >= 0 && gx < W)
      v = X[((size_t)gy * W + gx) * 64 + ic];
    patch[dy][xx][ic] = v;
  }
}

__global__ void mpft_conv3x3(const h16* __restrict__ X, const h16* __restrict__ Wp,
                             const float* __restrict__ bias, float* __restrict__ Y,
                             int H, int W, int OC, int relu){
  __shared__ h16 patch[3][18][64];   // sole LDS object: LDS offset 0
  const int strip = blockIdx.x;
  const int wtiles = W >> 4;
  const int y  = strip / wtiles;
  const int x0 = (strip % wtiles) << 4;
  const int HWl = H * W;

  const bool interior = (y >= 1) && (y <= H - 2) && (x0 >= 16) && (x0 + 17 <= W - 1);
#if USE_TDM
  if (interior) {
    if (threadIdx.x < 32) {  // one wave issues the tensor DMA
      unsigned long long ga =
          (unsigned long long)(const void*)(X + ((size_t)(y - 1) * W + (x0 - 1)) * 64);
      // D# group0: count=1, lds_addr=0 (patch), global_addr, type=2
      u32x4 g0;
      g0[0] = 1u;
      g0[1] = 0u;
      g0[2] = (unsigned)(ga & 0xFFFFFFFFu);
      g0[3] = (unsigned)((ga >> 32) & 0x01FFFFFFu) | (2u << 30);
      // D# group1: data_size=1 (2B); tensor dims/strides; tile 64x18x3
      unsigned td0 = 64u, td1 = (unsigned)W;
      unsigned tl0 = 64u, tl1 = 18u, tl2 = 3u;
      unsigned long long s0 = 64ull, s1 = (unsigned long long)W * 64ull;
      i32x8 g1;
      g1[0] = (int)(1u << 16);
      g1[1] = (int)((td0 & 0xFFFFu) << 16);
      g1[2] = (int)((td0 >> 16) | ((td1 & 0xFFFFu) << 16));
      g1[3] = (int)((td1 >> 16) | (tl0 << 16));
      g1[4] = (int)(tl1 | (tl2 << 16));
      g1[5] = (int)(s0 & 0xFFFFFFFFull);
      g1[6] = (int)((unsigned)(s0 >> 32) | ((unsigned)(s1 & 0xFFFFull) << 16));
      g1[7] = (int)(s1 >> 16);
      i32x4 g2; g2[0] = H; g2[1] = 0; g2[2] = 0; g2[3] = 0;
      i32x4 g3 = {0, 0, 0, 0};
#if __clang_major__ >= 23
      i32x8 g4 = {0, 0, 0, 0, 0, 0, 0, 0};
      __builtin_amdgcn_tensor_load_to_lds(g0, g1, g2, g3, g4, 0);
#else
      __builtin_amdgcn_tensor_load_to_lds(g0, g1, g2, g3, 0);
#endif
      __builtin_amdgcn_s_wait_tensorcnt(0);
    }
  } else {
    conv_stage_manual(X, patch, y, x0, H, W);
  }
#else
  (void)interior;
  conv_stage_manual(X, patch, y, x0, H, W);
#endif
  __syncthreads();

  const int lane = threadIdx.x & 31;
  const int wave = threadIdx.x >> 5;
  const int oBase = (blockIdx.y * 4 + wave) * 16;
  const int m = lane & 15;
  const int kOffA = (lane < 16) ? 0 : 8;
  const int kSelB = (lane < 16) ? 0 : 8;
  const int ocol = oBase + (lane & 15);
  v8f acc = {0.f,0.f,0.f,0.f,0.f,0.f,0.f,0.f};
  #pragma unroll
  for (int ks = 0; ks < 18; ++ks) {
    const int tap = ks >> 1;
    const int iBase = (ks & 1) * 32;
    const int dy = tap / 3, dx = tap % 3;
    const h16* pp = &patch[dy][m + dx][iBase + kOffA];
    v16h a = cat8_(*(const v8h*)pp, *(const v8h*)(pp + 16));
    const h16* bp = Wp + ((size_t)ks * OC + ocol) * 32 + kSelB;
    v16h b = cat8_(*(const v8h*)bp, *(const v8h*)(bp + 16));
    acc = __builtin_amdgcn_wmma_f32_16x16x32_f16(false, a, false, b,
                                                 (short)0, acc, false, false);
  }
  const int mAdd = (lane < 16) ? 0 : 8;
  #pragma unroll
  for (int r = 0; r < 8; ++r) {
    float v = acc[r] + bias[ocol];
    if (relu) v = fmaxf(v, 0.f);
    Y[(size_t)ocol * HWl + y * W + x0 + r + mAdd] = v;
  }
}

// pack OIHW f32 (IC=64) -> fragment-packed f16 [((ks*OC)+o)*32 + kloc]
__global__ void mpft_pack_conv(const float* __restrict__ w, h16* __restrict__ wp, int OC){
  int t = blockIdx.x * blockDim.x + threadIdx.x;
  if (t >= OC * 576) return;
  int kl = t & 31, r = t >> 5, o = r % OC, ks = r / OC;
  int tap = ks >> 1, ic = (ks & 1) * 32 + kl;
  wp[t] = (h16)w[(size_t)(o * 64 + ic) * 9 + tap];
}

// ---------------- elementwise / small kernels ----------------
__global__ void mpft_f2h(h16* d, const float* s, int n){
  int t = blockIdx.x * blockDim.x + threadIdx.x; if (t < n) d[t] = (h16)s[t];
}
// CHW f32 -> HWC f16 (C=64) for conv activation staging
__global__ void mpft_f2h_hwc(const float* __restrict__ s, h16* __restrict__ d, int HWl){
  int t = blockIdx.x * blockDim.x + threadIdx.x; if (t >= 64 * HWl) return;
  int c = t / HWl, p = t % HWl;
  d[(size_t)p * 64 + c] = (h16)s[t];
}
__global__ void mpft_copy(float* d, const float* s, int n){
  int t = blockIdx.x * blockDim.x + threadIdx.x; if (t < n) d[t] = s[t];
}
__global__ void mpft_add(float* d, const float* s, int n){
  int t = blockIdx.x * blockDim.x + threadIdx.x; if (t < n) d[t] += s[t];
}
__global__ void mpft_addbias(float* d, const float* s, const float* b, int N, int n){
  int t = blockIdx.x * blockDim.x + threadIdx.x; if (t < n) d[t] += s[t] + b[t % N];
}
__global__ void mpft_bias_gelu(float* d, const float* b, int N, int n){
  int t = blockIdx.x * blockDim.x + threadIdx.x; if (t < n) d[t] = geluf_(d[t] + b[t % N]);
}
// shallow conv: 1 input channel, relu
__global__ void mpft_conv1in(const float* __restrict__ x, const float* __restrict__ w,
                             const float* __restrict__ b, float* __restrict__ y){
  int t = blockIdx.x * blockDim.x + threadIdx.x;
  if (t >= 64 * HW) return;
  int o = t >> 14, p = t & (HW-1), yy = p >> 7, xx = p & 127;
  float s = b[o];
  #pragma unroll
  for (int ky = 0; ky < 3; ++ky) {
    int gy = yy + ky - 1; if (gy < 0 || gy > 127) continue;
    #pragma unroll
    for (int kx = 0; kx < 3; ++kx) {
      int gx = xx + kx - 1; if (gx < 0 || gx > 127) continue;
      s += w[o * 9 + ky * 3 + kx] * x[gy * 128 + gx];
    }
  }
  y[t] = fmaxf(s, 0.f);
}
// layernorm over last dim 64
__global__ void mpft_ln64(const float* __restrict__ xin, const float* __restrict__ w,
                          const float* __restrict__ b, float* __restrict__ y, int rows){
  int r = blockIdx.x * blockDim.x + threadIdx.x; if (r >= rows) return;
  const float* p = xin + (size_t)r * 64;
  float mu = 0.f;
  for (int c = 0; c < 64; ++c) mu += p[c];
  mu *= (1.f/64.f);
  float v = 0.f;
  for (int c = 0; c < 64; ++c){ float d = p[c]-mu; v += d*d; }
  v *= (1.f/64.f);
  float inv = rsqrtf(v + 1e-5f);
  float* q = y + (size_t)r * 64;
  for (int c = 0; c < 64; ++c) q[c] = (p[c]-mu) * inv * w[c] + b[c];
}
__global__ void mpft_chw2hwc(const float* s, float* d){
  int t = blockIdx.x * blockDim.x + threadIdx.x; if (t >= 64*HW) return;
  int c = t / HW, p = t % HW; d[(size_t)p * 64 + c] = s[t];
}
__global__ void mpft_hwc2chw(const float* s, float* d){
  int t = blockIdx.x * blockDim.x + threadIdx.x; if (t >= 64*HW) return;
  int c = t / HW, p = t % HW; d[t] = s[(size_t)p * 64 + c];
}
// ---- CBAM ----
__global__ void mpft_chan_reduce(const float* __restrict__ X, float* avg, float* mx){
  __shared__ float ss[256], sm[256];
  int c = blockIdx.x; float s = 0.f, m = -1e30f;
  for (int p = threadIdx.x; p < HW; p += 256){ float v = X[(size_t)c*HW + p]; s += v; m = fmaxf(m, v); }
  ss[threadIdx.x] = s; sm[threadIdx.x] = m; __syncthreads();
  for (int st = 128; st > 0; st >>= 1){
    if (threadIdx.x < st){ ss[threadIdx.x] += ss[threadIdx.x+st];
                           sm[threadIdx.x] = fmaxf(sm[threadIdx.x], sm[threadIdx.x+st]); }
    __syncthreads();
  }
  if (threadIdx.x == 0){ avg[c] = ss[0] * (1.f/HW); mx[c] = sm[0]; }
}
__global__ void mpft_ca_fc(const float* avg, const float* mx, const float* w1,
                           const float* w2, float* catt){
  __shared__ float t1a[4], t1m[4];
  int t = threadIdx.x;
  if (t < 4){
    float sa = 0.f, sm = 0.f;
    for (int c = 0; c < 64; ++c){ sa += w1[t*64+c]*avg[c]; sm += w1[t*64+c]*mx[c]; }
    t1a[t] = fmaxf(sa, 0.f); t1m[t] = fmaxf(sm, 0.f);
  }
  __syncthreads();
  float ya = 0.f, ym = 0.f;
  for (int j = 0; j < 4; ++j){ ya += w2[t*4+j]*t1a[j]; ym += w2[t*4+j]*t1m[j]; }
  catt[t] = sigmoidf_(ya + ym);
}
__global__ void mpft_scale_chan(float* d, const float* s, const float* catt){
  int t = blockIdx.x * blockDim.x + threadIdx.x; if (t >= 64*HW) return;
  d[t] = s[t] * catt[t >> 14];
}
__global__ void mpft_spatial_maps(const float* __restrict__ X, float* sa0, float* sa1){
  int p = blockIdx.x * blockDim.x + threadIdx.x; if (p >= HW) return;
  float s = 0.f, m = -1e30f;
  for (int c = 0; c < 64; ++c){ float v = X[(size_t)c*HW + p]; s += v; m = fmaxf(m, v); }
  sa0[p] = s * (1.f/64.f); sa1[p] = m;
}
__global__ void mpft_sa_conv(const float* sa0, const float* sa1, const float* w, float* smap){
  int p = blockIdx.x * blockDim.x + threadIdx.x; if (p >= HW) return;
  int y = p >> 7, x = p & 127; float s = 0.f;
  for (int c2 = 0; c2 < 2; ++c2){
    const float* mp = c2 ? sa1 : sa0;
    for (int ky = 0; ky < 7; ++ky){
      int gy = y + ky - 3; if (gy < 0 || gy > 127) continue;
      for (int kx = 0; kx < 7; ++kx){
        int gx = x + kx - 3; if (gx < 0 || gx > 127) continue;
        s += w[c2*49 + ky*7 + kx] * mp[gy*128 + gx];
      }
    }
  }
  smap[p] = sigmoidf_(s);
}
__global__ void mpft_mul_smap(float* d, const float* smap){
  int t = blockIdx.x * blockDim.x + threadIdx.x; if (t >= 64*HW) return;
  d[t] *= smap[t & (HW-1)];
}
// ---- Mamba pieces ----
__global__ void mpft_xconv(const float* __restrict__ xz, const float* __restrict__ cc,
                           float* __restrict__ out){
  int t = blockIdx.x * blockDim.x + threadIdx.x; if (t >= HW*DI) return;
  int p = t >> 7, d = t & 127;
  out[t] = siluf_(xz[(size_t)p*XZW + d] * cc[d]);
}
// selective scan along a line of 128 steps; block = 128 (d), grid = 128 (lines)
__global__ void mpft_scan(const float* __restrict__ xz, const float* __restrict__ xc,
                          const float* __restrict__ Alog, const float* __restrict__ Dp,
                          float* __restrict__ outcat, int baseStride, int step, int colOff){
  __shared__ float Bsh[128][DS], Csh[128][DS];
  const int d = threadIdx.x;
  const int base = blockIdx.x * baseStride;
  { // thread d stages time index d
    int p = base + d * step;
    for (int n = 0; n < DS; ++n){
      Bsh[d][n] = xz[(size_t)p*XZW + 2*DI + n];
      Csh[d][n] = xz[(size_t)p*XZW + 2*DI + DS + n];
    }
  }
  __syncthreads();
  float aA[DS], h[DS];
  #pragma unroll
  for (int n = 0; n < DS; ++n){ aA[n] = -__expf(Alog[d*DS + n]); h[n] = 0.f; }
  const float Dd = Dp[d];
  for (int t = 0; t < 128; ++t){
    int p = base + t * step;
    float ds = softplusf_(xz[(size_t)p*XZW + 2*DI + 2*DS + d]);
    float xv = xc[(size_t)p*DI + d];
    float dsx = ds * xv, yacc = 0.f;
    #pragma unroll
    for (int n = 0; n < DS; ++n){
      h[n] = __expf(ds * aA[n]) * h[n] + dsx * Bsh[t][n];
      yacc += h[n] * Csh[t][n];
    }
    outcat[(size_t)p*(2*DI) + colOff + d] = yacc + xv * Dd;
  }
}
__global__ void mpft_silu_gate(float* __restrict__ y, const float* __restrict__ xz){
  int t = blockIdx.x * blockDim.x + threadIdx.x; if (t >= HW*DI) return;
  int p = t >> 7, d = t & 127;
  y[t] *= siluf_(xz[(size_t)p*XZW + DI + d]);
}
// ---- fusion / tail ----
// build fragment-packed f16 B operand [(kb*HW+p)*32 + kloc] from local||glob (128 ch, CHW)
__global__ void mpft_concat_f16(const float* loc, const float* glb, h16* d){
  int t = blockIdx.x * blockDim.x + threadIdx.x; if (t >= 128*HW) return;
  int c = t / HW, p = t % HW;
  float v = (c < 64) ? loc[(size_t)c*HW + p] : glb[(size_t)(c-64)*HW + p];
  d[((size_t)(c >> 5) * HW + p) * 32 + (c & 31)] = (h16)v;
}
__global__ void mpft_gatemix(float* out, const float* gpre, const float* glb, const float* loc, int n){
  int t = blockIdx.x * blockDim.x + threadIdx.x; if (t >= n) return;
  float g = sigmoidf_(gpre[t]);
  out[t] = g * glb[t] + (1.f - g) * loc[t];
}
__global__ void mpft_pixshuf(const float* __restrict__ s, float* __restrict__ d){
  int t = blockIdx.x * blockDim.x + threadIdx.x; if (t >= 64*65536) return;
  int c = t / 65536, r = t % 65536;
  int oy = r >> 8, ox = r & 255;
  int y = oy >> 1, ry = oy & 1, x = ox >> 1, rx = ox & 1;
  d[t] = s[(size_t)(c*4 + ry*2 + rx) * HW + y*128 + x];
}
__global__ void mpft_rec2(const float* __restrict__ X, const float* __restrict__ w,
                          const float* __restrict__ b, float* __restrict__ out){
  int p = blockIdx.x * blockDim.x + threadIdx.x; if (p >= 65536) return;
  int y = p >> 8, x = p & 255; float s = b[0];
  for (int ic = 0; ic < 64; ++ic){
    const float* xp = X + (size_t)ic * 65536;
    for (int ky = 0; ky < 3; ++ky){
      int gy = y + ky - 1; if (gy < 0 || gy > 255) continue;
      for (int kx = 0; kx < 3; ++kx){
        int gx = x + kx - 1; if (gx < 0 || gx > 255) continue;
        s += w[ic*9 + ky*3 + kx] * xp[gy*256 + gx];
      }
    }
  }
  out[p] = s;
}
__global__ void mpft_bicubic_add(const float* __restrict__ x, const float* __restrict__ rec,
                                 float* __restrict__ out){
  int p = blockIdx.x * blockDim.x + threadIdx.x; if (p >= 65536) return;
  int oy = p >> 8, ox = p & 255;
  float sy = (oy + 0.5f) * 0.5f - 0.5f, sx = (ox + 0.5f) * 0.5f - 0.5f;
  int iy = (int)floorf(sy), ix = (int)floorf(sx);
  float ty = sy - iy, tx = sx - ix;
  float wy[4], wx[4];
  #pragma unroll
  for (int j = 0; j < 4; ++j){ wy[j] = cubicw_(ty - (j-1)); wx[j] = cubicw_(tx - (j-1)); }
  float s = 0.f;
  #pragma unroll
  for (int j = 0; j < 4; ++j){
    int gy = min(max(iy - 1 + j, 0), 127); float rs = 0.f;
    #pragma unroll
    for (int i = 0; i < 4; ++i){
      int gx = min(max(ix - 1 + i, 0), 127);
      rs += wx[i] * x[gy*128 + gx];
    }
    s += wy[j] * rs;
  }
  out[p] = rec[p] + s;
}

// ---------------- host ----------------
extern "C" void kernel_launch(void* const* d_in, const int* in_sizes, int n_in,
                              void* d_out, int out_size, void* d_ws, size_t ws_size,
                              hipStream_t stream) {
  (void)in_sizes; (void)n_in; (void)out_size; (void)ws_size;
  auto IN = [&](int i){ return (const float*)d_in[i]; };
  const float* x = IN(0);

  // ---- workspace carve ----
  char* w8 = (char*)d_ws; size_t off = 0;
  auto carve = [&](size_t bytes)->char*{ char* p = w8 + off; off += (bytes + 255) & ~(size_t)255; return p; };
  float* Af  = (float*)carve((size_t)64*HW*4);
  float* Bf  = (float*)carve((size_t)64*HW*4);
  float* Cf  = (float*)carve((size_t)64*HW*4);
  float* Df  = (float*)carve((size_t)64*HW*4);          // local
  h16*   X16 = (h16*)  carve((size_t)64*65536*2);       // f16 activations, HWC (max 64ch@256^2)
  h16*   WPh = (h16*)  carve((size_t)576*256*2);        // packed conv weights
  float* M1f = (float*)carve((size_t)HW*XZW*4);         // xz / 256ch conv out / rec1
  float* M2f = (float*)carve((size_t)64*65536*4);       // cat / shuffled / gate out
  float* M3f = (float*)carve((size_t)HW*DI*4);          // xconv / y1 / SA maps / rec2
  h16*   M4h = (h16*)  carve((size_t)HW*256*2);         // f16 multiuse (A operands, concat)
  float* M5a = (float*)carve((size_t)HW*64*4);          // mamba stream (HWC)
  float* M5b = (float*)carve((size_t)HW*64*4);          // ln buffer
  h16*   W16 = (h16*)  carve((size_t)32768*2);          // packed f16 small weights
  float* SMavg = (float*)carve(64*4);
  float* SMmx  = (float*)carve(64*4);
  float* SMcat = (float*)carve(64*4);
  float* SA0 = M3f, *SA1 = M3f + HW, *SMAP = M3f + 2*HW;

  auto EW = [&](int n){ return dim3((n + 255) / 256); };
  auto f2hA = [&](const float* s, h16* d, int HWl){   // CHW f32 -> HWC f16 (conv input)
    mpft_f2h_hwc<<<EW(64*HWl),256,0,stream>>>(s, d, HWl);
  };
  auto conv = [&](const h16* Xa, const float* wO, const float* bias, float* Y,
                  int Hs, int Ws, int OC, int relu){
    mpft_pack_conv<<<EW(OC*576),256,0,stream>>>(wO, WPh, OC);
    mpft_conv3x3<<<dim3(Hs*Ws/16, OC/64),128,0,stream>>>(Xa, WPh, bias, Y, Hs, Ws, OC, relu);
  };
  auto gemmW = [&](const h16* Aa, const float* Bw, float* Cc, int M, int N, int K){
    mpft_pack_b<<<EW(K*N),256,0,stream>>>(Bw, W16, N, K*N);
    mpft_gemm<<<dim3(M/16, (N/16 + 3)/4),128,0,stream>>>(Aa, W16, Cc, M, N, K);
  };

  // ---- shallow ----
  mpft_conv1in<<<EW(64*HW),256,0,stream>>>(x, IN(1), IN(2), Af);
  f2hA(Af, X16, HW);
  conv(X16, IN(3), IN(4), Af, 128, 128, 64, 0);

  // ---- PFT blocks ----
  for (int b = 0; b < 3; ++b){
    int base = 5 + b*17;
    mpft_copy<<<EW(64*HW),256,0,stream>>>(Cf, Af, 64*HW);      // block input
    for (int r = 0; r < 3; ++r){
      int rb = base + r*4;
      f2hA(Af, X16, HW);
      conv(X16, IN(rb), IN(rb+1), Bf, 128, 128, 64, 1);
      f2hA(Bf, X16, HW);
      conv(X16, IN(rb+2), IN(rb+3), Bf, 128, 128, 64, 0);
      mpft_add<<<EW(64*HW),256,0,stream>>>(Af, Bf, 64*HW);
    }
    // CBAM
    mpft_chan_reduce<<<64,256,0,stream>>>(Af, SMavg, SMmx);
    mpft_ca_fc<<<1,64,0,stream>>>(SMavg, SMmx, IN(base+12), IN(base+13), SMcat);
    mpft_scale_chan<<<EW(64*HW),256,0,stream>>>(M1f, Af, SMcat);
    mpft_spatial_maps<<<EW(HW),256,0,stream>>>(M1f, SA0, SA1);
    mpft_sa_conv<<<EW(HW),256,0,stream>>>(SA0, SA1, IN(base+14), SMAP);
    mpft_mul_smap<<<EW(64*HW),256,0,stream>>>(M1f, SMAP);
    mpft_add<<<EW(64*HW),256,0,stream>>>(M1f, Cf, 64*HW);      // + block input
    f2hA(M1f, X16, HW);
    conv(X16, IN(base+15), IN(base+16), Af, 128, 128, 64, 0);  // fuse
  }

  // ---- Mamba global branch ----
  mpft_copy<<<EW(64*HW),256,0,stream>>>(Df, Af, 64*HW);        // local
  mpft_chw2hwc<<<EW(64*HW),256,0,stream>>>(Af, M5a);
  for (int m = 0; m < 2; ++m){
    int base = 56 + m*14;
    mpft_ln64<<<EW(HW),256,0,stream>>>(M5a, IN(base), IN(base+1), M5b, HW);
    mpft_f2h<<<EW(HW*64),256,0,stream>>>(M4h, M5b, HW*64);
    gemmW(M4h, IN(base+2), M1f, HW, XZW, 64);                  // xz = ln @ in_proj
    mpft_xconv<<<EW(HW*DI),256,0,stream>>>(M1f, IN(base+3), M3f);
    mpft_scan<<<128,128,0,stream>>>(M1f, M3f, IN(base+4), IN(base+5), M2f, 128, 1,   0);
    mpft_scan<<<128,128,0,stream>>>(M1f, M3f, IN(base+4), IN(base+5), M2f, 1,   128, DI);
    mpft_f2h<<<EW(HW*2*DI),256,0,stream>>>(M4h, M2f, HW*2*DI);
    gemmW(M4h, IN(base+6), M3f, HW, DI, 2*DI);                 // @ dir_w
    mpft_silu_gate<<<EW(HW*DI),256,0,stream>>>(M3f, M1f);      // * silu(zg)
    mpft_f2h<<<EW(HW*DI),256,0,stream>>>(M4h, M3f, HW*DI);
    gemmW(M4h, IN(base+7), M5b, HW, 64, DI);                   // @ out_w
    mpft_add<<<EW(HW*64),256,0,stream>>>(M5a, M5b, HW*64);     // residual
    mpft_ln64<<<EW(HW),256,0,stream>>>(M5a, IN(base+8), IN(base+9), M5b, HW);
    mpft_f2h<<<EW(HW*64),256,0,stream>>>(M4h, M5b, HW*64);
    gemmW(M4h, IN(base+10), M1f, HW, 256, 64);                 // mlp_w1
    mpft_bias_gelu<<<EW(HW*256),256,0,stream>>>(M1f, IN(base+11), 256, HW*256);
    mpft_f2h<<<EW(HW*256),256,0,stream>>>(M4h, M1f, HW*256);
    gemmW(M4h, IN(base+12), M5b, HW, 64, 256);                 // mlp_w2
    mpft_addbias<<<EW(HW*64),256,0,stream>>>(M5a, M5b, IN(base+13), 64, HW*64);
  }
  mpft_ln64<<<EW(HW),256,0,stream>>>(M5a, IN(84), IN(85), M5b, HW);
  mpft_hwc2chw<<<EW(64*HW),256,0,stream>>>(M5b, Bf);           // glob

  // ---- gated fusion (1x1 conv as GEMM: weights(A) x CHW activations(B, packed)) ----
  mpft_concat_f16<<<EW(128*HW),256,0,stream>>>(Df, Bf, M4h);
  mpft_f2h<<<EW(64*128),256,0,stream>>>(W16, IN(86), 64*128);  // A: row-major f16
  mpft_gemm<<<dim3(64/16, (HW/16 + 3)/4),128,0,stream>>>(W16, M4h, M2f, 64, HW, 128);
  mpft_gatemix<<<EW(64*HW),256,0,stream>>>(Af, M2f, Bf, Df, 64*HW);

  // ---- upsample + reconstruction ----
  f2hA(Af, X16, HW);
  conv(X16, IN(87), IN(88), M1f, 128, 128, 256, 0);            // up conv (64->256)
  mpft_pixshuf<<<EW(64*65536),256,0,stream>>>(M1f, M2f);
  f2hA(M2f, X16, 65536);
  conv(X16, IN(89), IN(90), M1f, 256, 256, 64, 1);             // rec conv1 + relu
  mpft_rec2<<<EW(65536),256,0,stream>>>(M1f, IN(91), IN(92), M3f);
  mpft_bicubic_add<<<EW(65536),256,0,stream>>>(x, M3f, (float*)d_out);
}
